// FrenetPathMultiTargetGCN_54296976556449
// MI455X (gfx1250) — matrix operand order repaired and checked by reading, not compiled, not run on hardware
//
#include <hip/hip_runtime.h>
#include <cstdint>

// ---------------- problem dims ----------------
#define NN   50000
#define DD   64
#define HH   4
#define CC   64
#define HCC  256      // H*C
#define EE   300000
#define PP   20000
#define EPN_ 300000
#define VV   2000
#define NAA  128
#define HID_ 96       // (NA+D)/2
#define KIN_ 192      // NA+D

// ---------------- WMMA types ----------------
typedef __attribute__((ext_vector_type(16))) __bf16 v16bf;
typedef __attribute__((ext_vector_type(8)))  float  v8f;

union BFrag {
  v16bf bf;
  unsigned short us[16];
};

__device__ inline v8f wmma_bf16(v16bf a, v16bf b, v8f c) {
  return __builtin_amdgcn_wmma_f32_16x16x32_bf16(false, a, false, b, (short)0, c,
                                                 false, false);
}

// fp32 -> bf16 bits (RNE)
__device__ inline unsigned short f2bf(float f) {
  unsigned u = __float_as_uint(f);
  u += 0x7fffu + ((u >> 16) & 1u);
  return (unsigned short)(u >> 16);
}
__device__ inline float bf2f(unsigned short s) {
  return __uint_as_float(((unsigned)s) << 16);
}

// monotonic int encoding of float for atomicMax-based segment max
__device__ inline int f2ord(float f) {
  int i = __float_as_int(f);
  return i >= 0 ? i : (i ^ 0x7fffffff);
}
__device__ inline float ord2f(int i) {
  return __int_as_float(i >= 0 ? i : (i ^ 0x7fffffff));
}

// ---------------- async global -> LDS staging (CDNA5 ASYNCcnt path) -------
#if __has_builtin(__builtin_amdgcn_global_load_async_to_lds_b128)
#define HAVE_ASYNC_LDS 1
#else
#define HAVE_ASYNC_LDS 0
#endif

typedef int v4i_ __attribute__((vector_size(16)));

// stage 16 bytes per lane from global into LDS
__device__ inline void stage16(void* lds, const void* g) {
#if HAVE_ASYNC_LDS
  // global pointer: flat address == global address (AS1, 64-bit); builtin
  // parameter is non-const, so cast constness away.
  __attribute__((address_space(1))) v4i_* gp =
      (__attribute__((address_space(1))) v4i_*)(uintptr_t)g;
  // LDS pointer: low 32 bits of the flat address are the LDS byte offset (AS3)
  __attribute__((address_space(3))) v4i_* lp =
      (__attribute__((address_space(3))) v4i_*)(unsigned)(uintptr_t)lds;
  __builtin_amdgcn_global_load_async_to_lds_b128(gp, lp, 0, 0);
#else
  *(float4*)lds = *(const float4*)g;
#endif
}

__device__ inline void stage_wait() {
#if HAVE_ASYNC_LDS
#if __has_builtin(__builtin_amdgcn_s_wait_asynccnt)
  __builtin_amdgcn_s_wait_asynccnt(0);
#else
  asm volatile("s_wait_asynccnt 0x0" ::: "memory");
#endif
#endif
}

// A fragment (16x32 bf16) built from an fp32 row-major LDS tile (ld floats).
// ISA layout: lanes 0-15 hold row m=lane, K = 0..7 & 16..23 ; lanes 16-31
// hold K = 8..15 & 24..31 (k0 = tile K offset).
__device__ inline v16bf load_a_frag_f32(const float* a, int ld, int lane, int k0) {
  int m = lane & 15;
  int g = (lane >> 4) << 3;  // 0 or 8
  const float* p = a + m * ld + k0 + g;
  BFrag f;
#pragma unroll
  for (int i = 0; i < 8; ++i) {
    f.us[i]     = f2bf(p[i]);
    f.us[i + 8] = f2bf(p[16 + i]);
  }
  return f.bf;
}

// Same but from a bf16 (ushort) LDS tile
__device__ inline v16bf load_a_frag(const unsigned short* a, int ld, int lane, int k0) {
  int m = lane & 15;
  int g = (lane >> 4) << 3;
  const unsigned short* p = a + m * ld + k0 + g;
  BFrag f;
#pragma unroll
  for (int i = 0; i < 8; ++i) {
    f.us[i]     = p[i];
    f.us[i + 8] = p[16 + i];
  }
  return f.bf;
}

// Pre-packed B fragment: contiguous [frag][lane][16 halves]
__device__ inline v16bf load_b_frag(const unsigned short* pw, int frag, int lane) {
  return reinterpret_cast<const v16bf*>(pw)[frag * 32 + lane];
}

// ---------------- init kernels ----------------
__global__ void fill_f32(float* p, float v, int n) {
  int i = blockIdx.x * blockDim.x + threadIdx.x;
  if (i < n) p[i] = v;
}
__global__ void fill_i32(int* p, int v, int n) {
  int i = blockIdx.x * blockDim.x + threadIdx.x;
  if (i < n) p[i] = v;
}

// ---------------- weight packer ----------------
// W is row-major [K][Nn] f32.  Packs bf16 B fragments in the ISA layout:
// lane n = lane&15 (column), g = lane>>4: K = 8g+0..7 then 16+8g+0..7.
// Fragment order: frag = nt*Kt + kt.
__global__ void pack_b_kernel(const float* W, unsigned short* dst, int K, int Nn) {
  int frag = blockIdx.x;
  int Kt = K >> 5;
  int nt = frag / Kt, kt = frag - nt * Kt;
  int lane = threadIdx.x;
  int n = nt * 16 + (lane & 15);
  int kbase = kt * 32 + ((lane >> 4) << 3);
  unsigned short* o = dst + (size_t)frag * 512 + lane * 16;
#pragma unroll
  for (int i = 0; i < 16; ++i)
    o[i] = f2bf(W[(size_t)(kbase + (i & 7) + ((i >> 3) << 4)) * Nn + n]);
}

// ---------------- QKV GEMM: x[N,64] @ Wqkv[64,768] -> qkv bf16 [N,768] ----
__global__ void qkv_gemm_kernel(const float* x, const unsigned short* pw_qkv,
                                const float* bq, const float* bk, const float* bv,
                                unsigned short* qkv) {
  __shared__ float As[16 * 64];
  int t = threadIdx.x, lane = t & 31, wid = t >> 5;
  size_t row0 = (size_t)blockIdx.x * 16;
  stage16(As + t * 4, x + row0 * 64 + t * 4);  // 16x64 f32 tile via async DMA
  stage_wait();
  __syncthreads();
  v16bf a0 = load_a_frag_f32(As, 64, lane, 0);
  v16bf a1 = load_a_frag_f32(As, 64, lane, 32);
  int n = lane & 15, rb = (lane >> 4) * 8;
#pragma unroll
  for (int ti = 0; ti < 6; ++ti) {
    int nt = wid * 6 + ti;  // 48 tiles over 768 cols
    v8f acc = {};
    acc = wmma_bf16(a0, load_b_frag(pw_qkv, nt * 2 + 0, lane), acc);
    acc = wmma_bf16(a1, load_b_frag(pw_qkv, nt * 2 + 1, lane), acc);
    int col = nt * 16 + n;
    float bias = (col < 256) ? bq[col] : (col < 512 ? bk[col - 256] : bv[col - 512]);
#pragma unroll
    for (int j = 0; j < 8; ++j)
      qkv[(row0 + rb + j) * 768 + col] = f2bf(acc[j] + bias);
  }
}

// ---------------- edge attention ----------------
__global__ void edge_alpha_kernel(const unsigned short* qkv, const int* ei,
                                  float* alpha, int* segmax) {
  int idx = blockIdx.x * blockDim.x + threadIdx.x;
  if (idx >= EE * HH) return;
  int e = idx >> 2, h = idx & 3;
  int src = ei[e], dst = ei[EE + e];
  const unsigned* qp = (const unsigned*)(qkv + (size_t)dst * 768 + h * 64);
  const unsigned* kp = (const unsigned*)(qkv + (size_t)src * 768 + 256 + h * 64);
  __builtin_prefetch(qp, 0, 0);
  __builtin_prefetch(kp, 0, 0);
  float s = 0.f;
#pragma unroll
  for (int w = 0; w < 32; ++w) {
    unsigned uq = qp[w], uk = kp[w];
    s += __uint_as_float(uq << 16) * __uint_as_float(uk << 16);
    s += __uint_as_float(uq & 0xffff0000u) * __uint_as_float(uk & 0xffff0000u);
  }
  float a = s * 0.125f;  // 1/sqrt(64)
  alpha[idx] = a;
  atomicMax(segmax + dst * HH + h, f2ord(a));
}

__global__ void edge_exp_kernel(const int* ei, float* alpha, const int* segmax,
                                float* segsum) {
  int idx = blockIdx.x * blockDim.x + threadIdx.x;
  if (idx >= EE * HH) return;
  int e = idx >> 2, h = idx & 3;
  int dst = ei[EE + e];
  float ex = __expf(alpha[idx] - ord2f(segmax[dst * HH + h]));
  alpha[idx] = ex;
  atomicAdd(segsum + dst * HH + h, ex);
}

__global__ void edge_scatter_kernel(const unsigned short* qkv, const int* ei,
                                    const float* alpha, const float* segsum,
                                    float* attn) {
  int t = blockIdx.x * blockDim.x + threadIdx.x;
  if (t >= EE * 64) return;
  int e = t >> 6, d = t & 63;
  int src = ei[e], dst = ei[EE + e];
#pragma unroll
  for (int h = 0; h < HH; ++h) {
    float w = alpha[e * HH + h] / (segsum[dst * HH + h] + 1e-16f);
    float v = bf2f(qkv[(size_t)src * 768 + 512 + h * 64 + d]);
    atomicAdd(attn + (size_t)dst * 256 + h * 64 + d, w * v);
  }
}

// ---------------- fused node post: skip GEMM + LN1 + FFN + LN2 ------------
__global__ void node_post_kernel(const float* attn, const float* x,
                                 const unsigned short* pw_skip, const float* bskip,
                                 const float* ln1g, const float* ln1b,
                                 const unsigned short* pw_f1, const float* bf1_,
                                 const unsigned short* pw_f2, const float* bf2_,
                                 const float* ln2g, const float* ln2b,
                                 float* out2) {
  __shared__ float A1[16 * 256];          // attn tile f32 (async staged)
  __shared__ float xs[16 * 64];           // x tile f32 (async staged)
  __shared__ float h1[16 * 64];
  __shared__ unsigned short A2[16 * 64];
  __shared__ unsigned short FFs[16 * 128];
  __shared__ float mu[16], rs[16];
  int t = threadIdx.x, lane = t & 31, wid = t >> 5;
  size_t row0 = (size_t)blockIdx.x * 16;

#pragma unroll
  for (int j = 0; j < 4; ++j) {
    int i = (j * 256 + t) * 4;
    stage16(A1 + i, attn + row0 * 256 + i);
  }
  stage16(xs + t * 4, x + row0 * 64 + t * 4);
  stage_wait();
  __syncthreads();

  // skip GEMM (256 -> 64), waves 0..3
  if (wid < 4) {
    v8f acc = {};
#pragma unroll
    for (int kt = 0; kt < 8; ++kt)
      acc = wmma_bf16(load_a_frag_f32(A1, 256, lane, kt * 32),
                      load_b_frag(pw_skip, wid * 8 + kt, lane), acc);
    int n = lane & 15, rb = (lane >> 4) * 8, col = wid * 16 + n;
    float bias = bskip[col];
#pragma unroll
    for (int j = 0; j < 8; ++j) {
      int r = rb + j;
      h1[r * 64 + col] = acc[j] + bias + xs[r * 64 + col];
    }
  }
  __syncthreads();

  // LN1
  if (t < 16) {
    float s = 0.f, ss = 0.f;
    for (int c = 0; c < 64; ++c) {
      float v = h1[t * 64 + c];
      s += v;
      ss += v * v;
    }
    float m = s * (1.f / 64.f);
    mu[t] = m;
    rs[t] = rsqrtf(ss * (1.f / 64.f) - m * m + 1e-5f);
  }
  __syncthreads();
#pragma unroll
  for (int j = 0; j < 4; ++j) {
    int i = j * 256 + t, r = i >> 6, c = i & 63;
    A2[i] = f2bf((h1[i] - mu[r]) * rs[r] * ln1g[c] + ln1b[c]);
  }
  __syncthreads();

  // FF1 (64 -> 128), all 8 waves
  {
    v8f acc = {};
#pragma unroll
    for (int kt = 0; kt < 2; ++kt)
      acc = wmma_bf16(load_a_frag(A2, 64, lane, kt * 32),
                      load_b_frag(pw_f1, wid * 2 + kt, lane), acc);
    int n = lane & 15, rb = (lane >> 4) * 8, col = wid * 16 + n;
    float bias = bf1_[col];
#pragma unroll
    for (int j = 0; j < 8; ++j) {
      float v = acc[j] + bias;
      FFs[(rb + j) * 128 + col] = f2bf(v > 0.f ? v : 0.f);
    }
  }
  __syncthreads();

  // FF2 (128 -> 64), waves 0..3, residual = x
  if (wid < 4) {
    v8f acc = {};
#pragma unroll
    for (int kt = 0; kt < 4; ++kt)
      acc = wmma_bf16(load_a_frag(FFs, 128, lane, kt * 32),
                      load_b_frag(pw_f2, wid * 4 + kt, lane), acc);
    int n = lane & 15, rb = (lane >> 4) * 8, col = wid * 16 + n;
    float bias = bf2_[col];
#pragma unroll
    for (int j = 0; j < 8; ++j) {
      int r = rb + j;
      h1[r * 64 + col] = acc[j] + bias + xs[r * 64 + col];
    }
  }
  __syncthreads();

  // LN2 -> out2
  if (t < 16) {
    float s = 0.f, ss = 0.f;
    for (int c = 0; c < 64; ++c) {
      float v = h1[t * 64 + c];
      s += v;
      ss += v * v;
    }
    float m = s * (1.f / 64.f);
    mu[t] = m;
    rs[t] = rsqrtf(ss * (1.f / 64.f) - m * m + 1e-5f);
  }
  __syncthreads();
#pragma unroll
  for (int j = 0; j < 4; ++j) {
    int i = j * 256 + t, r = i >> 6, c = i & 63;
    out2[row0 * 64 + i] = (h1[i] - mu[r]) * rs[r] * ln2g[c] + ln2b[c];
  }
}

// ---------------- path aggregation ----------------
__global__ void path_agg_kernel(const int* pnu, const int* pnv, const float* out2,
                                float* psum, float* cnt) {
  int t = blockIdx.x * blockDim.x + threadIdx.x;
  if (t >= EPN_ * 64) return;
  int e = t >> 6, d = t & 63;
  int p = pnu[e], nd = pnv[e];
  atomicAdd(psum + (size_t)p * 64 + d, out2[(size_t)nd * 64 + d]);
  if (d == 0) atomicAdd(cnt + p, 1.0f);
}

// ---------------- PathPredNet: [paths|actors] @ Wp1 -> GN -> relu -> Wp2 ---
__global__ void path_pred_kernel(const float* psum, const float* cnt,
                                 const int* vpu, const float* actors,
                                 const unsigned short* pw_p1, const float* bp1,
                                 const float* gng, const float* gnb,
                                 const float* Wp2, const float* bp2,
                                 float* logits, int* vehmax) {
  __shared__ unsigned short As[32 * KIN_];
  __shared__ float Hs[32 * HID_];
  __shared__ int uu[32];
  __shared__ float rinv[32];
  int t = threadIdx.x, lane = t & 31, wid = t >> 5;
  int p0 = blockIdx.x * 32;

  if (t < 32) {
    int p = p0 + t;
    uu[t] = vpu[p];
    float c = cnt[p];
    rinv[t] = 1.0f / (c > 1.0f ? c : 1.0f);
  }
  __syncthreads();
  for (int i = t; i < 32 * KIN_; i += 128) {
    int r = i / KIN_, c = i - r * KIN_;
    int p = p0 + r;
    float v = (c < 64) ? psum[(size_t)p * 64 + c] * rinv[r]
                       : actors[(size_t)uu[r] * NAA + (c - 64)];
    As[i] = f2bf(v);
  }
  __syncthreads();

  // GEMM 32x96, K=192 : 12 tiles over 4 waves
#pragma unroll
  for (int jt = 0; jt < 3; ++jt) {
    int tt = wid + 4 * jt;
    int mt = tt / 6, nt = tt - mt * 6;
    v8f acc = {};
#pragma unroll
    for (int kt = 0; kt < 6; ++kt)
      acc = wmma_bf16(load_a_frag(As + mt * 16 * KIN_, KIN_, lane, kt * 32),
                      load_b_frag(pw_p1, nt * 6 + kt, lane), acc);
    int n = lane & 15, rb = (lane >> 4) * 8, col = nt * 16 + n;
    float bias = bp1[col];
#pragma unroll
    for (int j = 0; j < 8; ++j)
      Hs[(mt * 16 + rb + j) * HID_ + col] = acc[j] + bias;
  }
  __syncthreads();

  // GroupNorm(ng=1) + relu + final dot
  if (t < 32) {
    float s = 0.f, ss = 0.f;
    for (int c = 0; c < HID_; ++c) {
      float v = Hs[t * HID_ + c];
      s += v;
      ss += v * v;
    }
    float m = s * (1.f / HID_);
    float r = rsqrtf(ss * (1.f / HID_) - m * m + 1e-5f);
    float acc = 0.f;
    for (int c = 0; c < HID_; ++c) {
      float hn = (Hs[t * HID_ + c] - m) * r * gng[c] + gnb[c];
      if (hn > 0.f) acc += hn * Wp2[c];
    }
    float logit = acc + bp2[0];
    int p = p0 + t;
    logits[p] = logit;
    atomicMax(vehmax + uu[t], f2ord(logit));
  }
}

// ---------------- vehicle segment softmax ----------------
__global__ void veh_exp_kernel(const int* vpu, float* logits, const int* vehmax,
                               float* vehsum) {
  int p = blockIdx.x * blockDim.x + threadIdx.x;
  if (p >= PP) return;
  int u = vpu[p];
  float ex = __expf(logits[p] - ord2f(vehmax[u]));
  logits[p] = ex;
  atomicAdd(vehsum + u, ex);
}
__global__ void veh_norm_kernel(const int* vpu, const float* logits,
                                const float* vehsum, float* probs) {
  int p = blockIdx.x * blockDim.x + threadIdx.x;
  if (p >= PP) return;
  probs[p] = logits[p] / (vehsum[vpu[p]] + 1e-16f);
}

// ---------------- launcher ----------------
extern "C" void kernel_launch(void* const* d_in, const int* in_sizes, int n_in,
                              void* d_out, int out_size, void* d_ws, size_t ws_size,
                              hipStream_t stream) {
  (void)in_sizes; (void)n_in; (void)out_size; (void)ws_size;

  const float* x      = (const float*)d_in[0];
  const int*   ei     = (const int*)d_in[1];
  const int*   pnu    = (const int*)d_in[2];
  const int*   pnv    = (const int*)d_in[3];
  const int*   vpu    = (const int*)d_in[4];
  const float* actors = (const float*)d_in[5];
  const float* Wq = (const float*)d_in[6];  const float* bq = (const float*)d_in[7];
  const float* Wk = (const float*)d_in[8];  const float* bk = (const float*)d_in[9];
  const float* Wv = (const float*)d_in[10]; const float* bv = (const float*)d_in[11];
  const float* Wskip = (const float*)d_in[12]; const float* bskip = (const float*)d_in[13];
  const float* ln1g = (const float*)d_in[14]; const float* ln1b = (const float*)d_in[15];
  const float* Wf1 = (const float*)d_in[16]; const float* bf1_ = (const float*)d_in[17];
  const float* Wf2 = (const float*)d_in[18]; const float* bf2_ = (const float*)d_in[19];
  const float* ln2g = (const float*)d_in[20]; const float* ln2b = (const float*)d_in[21];
  const float* Wp1 = (const float*)d_in[22]; const float* bp1 = (const float*)d_in[23];
  const float* gng = (const float*)d_in[24]; const float* gnb = (const float*)d_in[25];
  const float* Wp2 = (const float*)d_in[26]; const float* bp2 = (const float*)d_in[27];

  float* out2  = (float*)d_out;
  float* probs = (float*)d_out + (size_t)NN * DD;

  // workspace carve-up (256B aligned)
  char* ws = (char*)d_ws;
  size_t off = 0;
  auto carve = [&](size_t bytes) {
    char* p = ws + off;
    off = (off + bytes + 255) & ~(size_t)255;
    return p;
  };
  unsigned short* qkv   = (unsigned short*)carve((size_t)NN * 768 * 2);
  float* attn           = (float*)carve((size_t)NN * HCC * 4);
  float* alpha          = (float*)carve((size_t)EE * HH * 4);
  int*   segmax         = (int*)carve((size_t)NN * HH * 4);
  float* segsum         = (float*)carve((size_t)NN * HH * 4);
  float* psum           = (float*)carve((size_t)PP * 64 * 4);
  float* cnt            = (float*)carve((size_t)PP * 4);
  float* logits         = (float*)carve((size_t)PP * 4);
  int*   vehmax         = (int*)carve((size_t)VV * 4);
  float* vehsum         = (float*)carve((size_t)VV * 4);
  unsigned short* pw_qkv  = (unsigned short*)carve(96 * 1024);  // 48 nt x 2 kt
  unsigned short* pw_skip = (unsigned short*)carve(32 * 1024);  // 4 nt x 8 kt
  unsigned short* pw_f1   = (unsigned short*)carve(16 * 1024);  // 8 nt x 2 kt
  unsigned short* pw_f2   = (unsigned short*)carve(16 * 1024);  // 4 nt x 4 kt
  unsigned short* pw_p1   = (unsigned short*)carve(36 * 1024);  // 6 nt x 6 kt

  const int TB = 256;
  auto blocks = [](long n, int tb) { return (int)((n + tb - 1) / tb); };

  // init accumulators (ws is poisoned by harness)
  fill_f32<<<blocks((long)NN * HCC, TB), TB, 0, stream>>>(attn, 0.f, NN * HCC);
  fill_f32<<<blocks((long)NN * HH, TB), TB, 0, stream>>>(segsum, 0.f, NN * HH);
  fill_f32<<<blocks((long)PP * 64, TB), TB, 0, stream>>>(psum, 0.f, PP * 64);
  fill_f32<<<blocks(PP, TB), TB, 0, stream>>>(cnt, 0.f, PP);
  fill_f32<<<blocks(VV, TB), TB, 0, stream>>>(vehsum, 0.f, VV);
  fill_i32<<<blocks((long)NN * HH, TB), TB, 0, stream>>>(segmax, (int)0x80000000, NN * HH);
  fill_i32<<<blocks(VV, TB), TB, 0, stream>>>(vehmax, (int)0x80000000, VV);

  // pack weights into WMMA B-fragment layout (bf16)
  pack_b_kernel<<<32, 32, 0, stream>>>(Wq, pw_qkv, 64, 256);
  pack_b_kernel<<<32, 32, 0, stream>>>(Wk, pw_qkv + 32 * 512, 64, 256);
  pack_b_kernel<<<32, 32, 0, stream>>>(Wv, pw_qkv + 64 * 512, 64, 256);
  pack_b_kernel<<<32, 32, 0, stream>>>(Wskip, pw_skip, 256, 64);
  pack_b_kernel<<<16, 32, 0, stream>>>(Wf1, pw_f1, 64, 128);
  pack_b_kernel<<<16, 32, 0, stream>>>(Wf2, pw_f2, 128, 64);
  pack_b_kernel<<<36, 32, 0, stream>>>(Wp1, pw_p1, 192, 96);

  // pipeline
  qkv_gemm_kernel<<<NN / 16, 256, 0, stream>>>(x, pw_qkv, bq, bk, bv, qkv);
  edge_alpha_kernel<<<blocks((long)EE * HH, TB), TB, 0, stream>>>(qkv, ei, alpha, segmax);
  edge_exp_kernel<<<blocks((long)EE * HH, TB), TB, 0, stream>>>(ei, alpha, segmax, segsum);
  edge_scatter_kernel<<<blocks((long)EE * 64, TB), TB, 0, stream>>>(qkv, ei, alpha, segsum, attn);
  node_post_kernel<<<NN / 16, 256, 0, stream>>>(attn, x, pw_skip, bskip, ln1g, ln1b,
                                                pw_f1, bf1_, pw_f2, bf2_, ln2g, ln2b, out2);
  path_agg_kernel<<<blocks((long)EPN_ * 64, TB), TB, 0, stream>>>(pnu, pnv, out2, psum, cnt);
  path_pred_kernel<<<PP / 32, 128, 0, stream>>>(psum, cnt, vpu, actors, pw_p1, bp1,
                                                gng, gnb, Wp2, bp2, logits, vehmax);
  veh_exp_kernel<<<blocks(PP, TB), TB, 0, stream>>>(vpu, logits, vehmax, vehsum);
  veh_norm_kernel<<<blocks(PP, TB), TB, 0, stream>>>(vpu, logits, vehsum, probs);
}
